// KA4GANC_86612310491765
// MI455X (gfx1250) — matrix operand order repaired and checked by reading, not compiled, not run on hardware
//
#include <hip/hip_runtime.h>
#include <hip/hip_bf16.h>

// ---------------------------------------------------------------------------
// KA4GANC forward for MI455X (gfx1250, wave32, WMMA + TDM).
// Big matmuls: bf16 WMMA (v_wmma_f32_16x16x32_bf16) with f32 accumulation.
// Attention GEMM stages the shared B operand (hcT) through LDS with the
// Tensor Data Mover (tensor_load_to_lds + s_wait_tensorcnt), double-buffered.
// Problem constants fixed by the reference: N=4096 nodes, IN=512, H=64,
// E=131072 edges.
// ---------------------------------------------------------------------------

typedef __attribute__((ext_vector_type(16))) __bf16 v16bf;
typedef __attribute__((ext_vector_type(8)))  __bf16 v8bf;
typedef __attribute__((ext_vector_type(8)))  float  v8f;
typedef __attribute__((ext_vector_type(4)))  unsigned int u32x4;
typedef __attribute__((ext_vector_type(8)))  int i32x8;
typedef __attribute__((ext_vector_type(4)))  int i32x4;

#define NODES 4096
#define INDIM 512
#define HDIM  64
#define TDM_KC 128   // K elements per TDM-staged B chunk (64 rows x 128 = 16KB)

__device__ __forceinline__ __bf16 f2bf(float f) {
  unsigned u = __builtin_bit_cast(unsigned, f);
  u += 0x7FFFu + ((u >> 16) & 1u);               // round-to-nearest-even
  unsigned short h = (unsigned short)(u >> 16);
  return __builtin_bit_cast(__bf16, h);
}

__device__ __forceinline__ v16bf pack16(v8bf lo, v8bf hi) {
  v16bf r;
#pragma unroll
  for (int e = 0; e < 8; ++e) { r[e] = lo[e]; r[e + 8] = hi[e]; }
  return r;
}

__global__ __launch_bounds__(256) void cvt_f32_bf16(const float* __restrict__ src,
                                                    __bf16* __restrict__ dst, int n) {
  int i = blockIdx.x * blockDim.x + threadIdx.x;
  if (i < n) dst[i] = f2bf(src[i]);
}

// ---------------------------------------------------------------------------
// Generic NT GEMM:  C[M,N] = act( A[M,K] @ B[N,K]^T + bias )
// A,B bf16 row-major, contiguous K. One wave computes a 16 x (16*NB) strip:
// the A fragment is loaded once and reused across NB WMMAs (2x flop/byte for
// NB=2). act: 0=none, 1=leaky(0.01), 2=elu
// ---------------------------------------------------------------------------
template <int NB>
__global__ __launch_bounds__(256) void wmma_gemm_nt(
    const __bf16* __restrict__ A, const __bf16* __restrict__ B,
    float* __restrict__ Cf, __bf16* __restrict__ Ch,
    const float* __restrict__ bias, int M, int N, int K, int act) {
  const int lane = threadIdx.x & 31;
  const int wave = threadIdx.x >> 5;
  const int ntn  = N / (16 * NB);
  int tile = blockIdx.x * (blockDim.x >> 5) + wave;
  int total = (M >> 4) * ntn;
  if (tile >= total) return;
  const int m0 = (tile / ntn) << 4;
  const int n0 = (tile % ntn) * (16 * NB);
  const int half = lane >> 4;     // which 16-lane half
  const int r    = lane & 15;

  const __bf16* arow = A + (size_t)(m0 + r) * K;
  v8f acc[NB];
#pragma unroll
  for (int b = 0; b < NB; ++b) acc[b] = (v8f){0.f, 0.f, 0.f, 0.f, 0.f, 0.f, 0.f, 0.f};

  for (int k0 = 0; k0 < K; k0 += 32) {
    // A 16x32 (MxK): lane half h holds K = {8h..8h+7} u {16+8h..16+8h+7}
    v16bf a = pack16(*(const v8bf*)(arow + k0 + half * 8),
                     *(const v8bf*)(arow + k0 + 16 + half * 8));
#pragma unroll
    for (int b = 0; b < NB; ++b) {
      // B 32x16 (KxN): lane half h holds K = {8h..8h+7} u {16+8h..16+8h+7}
      const __bf16* brow = B + (size_t)(n0 + 16 * b + r) * K;
      v16bf bb = pack16(*(const v8bf*)(brow + k0 + half * 8),
                        *(const v8bf*)(brow + k0 + 16 + half * 8));
      acc[b] = __builtin_amdgcn_wmma_f32_16x16x32_bf16(
          false, a, false, bb, (short)0, acc[b], false, false);
    }
  }

  // C/D layout: element v -> row m0 + v + 8*half, col n0 + r
#pragma unroll
  for (int b = 0; b < NB; ++b) {
#pragma unroll
    for (int v = 0; v < 8; ++v) {
      int row = m0 + v + 8 * half;
      int col = n0 + 16 * b + r;
      float val = acc[b][v];
      if (bias) val += bias[col];
      if (act == 1)      val = val > 0.f ? val : 0.01f * val;
      else if (act == 2) val = val > 0.f ? val : (__expf(val) - 1.f);
      size_t idx = (size_t)row * N + col;
      if (Cf) Cf[idx] = val;
      if (Ch) Ch[idx] = f2bf(val);
    }
  }
}

// ---------------------------------------------------------------------------
// TDM issue: stage B[0:64, kc:kc+TDM_KC] (row-major, row stride K bf16) into
// LDS at lds_off. D# built per CDNA5 ISA (2D tensor,
// data_size=2B, type=2 "image"); groups 2/3 unused (<=2D).
// ---------------------------------------------------------------------------
#if defined(__has_builtin)
#if __has_builtin(__builtin_amdgcn_tensor_load_to_lds)
#define HAVE_TDM 1
#endif
#endif

__device__ __forceinline__ void tdm_issue_b(const __bf16* gsrc, unsigned lds_off, int K) {
#ifdef HAVE_TDM
  unsigned long long ga = (unsigned long long)(size_t)gsrc;
  u32x4 g0;
  g0[0] = 1u;                                        // count=1, user mode
  g0[1] = lds_off;                                   // LDS byte address
  g0[2] = (unsigned)(ga & 0xFFFFFFFFull);            // global_addr[31:0]
  g0[3] = (unsigned)((ga >> 32) & 0x01FFFFFFull)     // global_addr[56:32]
          | (2u << 30);                              // type=2
  i32x8 g1;
  g1[0] = 0x00010000;                                // data_size=1 (2 bytes)
  g1[1] = (int)(((unsigned)K & 0xFFFFu) << 16);      // tensor_dim0 lo
  g1[2] = (int)(((unsigned)K >> 16) | (64u << 16));  // tensor_dim0 hi | tensor_dim1=64
  g1[3] = (int)((unsigned)TDM_KC << 16);             // tile_dim0 = 128
  g1[4] = 64;                                        // tile_dim1 = 64 rows
  g1[5] = K;                                         // tensor_dim0_stride lo32
  g1[6] = 0;
  g1[7] = 0;
  i32x4 z4 = {0, 0, 0, 0};
#if __clang_major__ >= 23
  i32x8 z8 = {0, 0, 0, 0, 0, 0, 0, 0};
  __builtin_amdgcn_tensor_load_to_lds(g0, g1, z4, z4, z8, 0);
#else
  __builtin_amdgcn_tensor_load_to_lds(g0, g1, z4, z4, 0);
#endif
#else
  (void)gsrc; (void)lds_off; (void)K;
#endif
}

// ---------------------------------------------------------------------------
// Attention GEMM with TDM-staged B:  hp[Mc,64] = elu( att[Mc,K] @ hcT[64,K]^T )
// Block tile 64(M) x 64(N); 8 waves in a 4x2 grid, each wave 16x32 (2 acc).
// B K-chunks (64x128 bf16 = 16KB) double-buffered in LDS via TDM; consumed
// with ds_load_b128 fragments. A streams straight from L2 (prefetched).
// ---------------------------------------------------------------------------
__global__ __launch_bounds__(256) void wmma_gemm_att_tdm(
    const __bf16* __restrict__ A, const __bf16* __restrict__ B,
    float* __restrict__ Cf, int K) {
  __shared__ __bf16 bstage[2][64 * TDM_KC];
  const int lane = threadIdx.x & 31;
  const int wave = threadIdx.x >> 5;
  const int half = lane >> 4, r = lane & 15;
  const int wm = wave >> 1, wn = wave & 1;           // 4(M) x 2(N) wave grid
  const int m0 = blockIdx.x * 64 + wm * 16;
  const int n0 = wn * 32;
  const __bf16* arow = A + (size_t)(m0 + r) * K;

  v8f acc0 = (v8f){0.f, 0.f, 0.f, 0.f, 0.f, 0.f, 0.f, 0.f};
  v8f acc1 = acc0;

  unsigned lds0 = (unsigned)(size_t)&bstage[0][0];
  unsigned lds1 = (unsigned)(size_t)&bstage[1][0];

#ifdef HAVE_TDM
  if (wave == 0) tdm_issue_b(B, lds0, K);
#endif

  int buf = 0;
  for (int kc = 0; kc < K; kc += TDM_KC) {
#ifdef HAVE_TDM
    if (wave == 0) {
      if (kc + TDM_KC < K) {
        tdm_issue_b(B + (size_t)kc + TDM_KC, buf ? lds0 : lds1, K);
        __builtin_amdgcn_s_wait_tensorcnt((short)1);   // current chunk done
      } else {
        __builtin_amdgcn_s_wait_tensorcnt((short)0);
      }
    }
#else
    // Fallback: cooperative global->LDS copy of B[0:64, kc:kc+TDM_KC)
    for (int t = threadIdx.x; t < 64 * (TDM_KC / 8); t += blockDim.x) {
      int row = t / (TDM_KC / 8), seg = t % (TDM_KC / 8);
      *(v8bf*)&bstage[buf][row * TDM_KC + seg * 8] =
          *(const v8bf*)(B + (size_t)row * K + kc + seg * 8);
    }
#endif
    __syncthreads();                                   // publish chunk to all waves
    if (kc + TDM_KC < K)
      __builtin_prefetch(arow + kc + TDM_KC, 0, 0);    // global_prefetch A stream
    const __bf16* bch = &bstage[buf][0];
#pragma unroll
    for (int ks = 0; ks < TDM_KC; ks += 32) {
      v16bf a = pack16(*(const v8bf*)(arow + kc + ks + half * 8),
                       *(const v8bf*)(arow + kc + ks + 16 + half * 8));
      const __bf16* b0 = bch + (size_t)(n0 + r) * TDM_KC + ks + half * 8;
      v16bf bb0 = pack16(*(const v8bf*)b0, *(const v8bf*)(b0 + 16));
      acc0 = __builtin_amdgcn_wmma_f32_16x16x32_bf16(
          false, a, false, bb0, (short)0, acc0, false, false);
      const __bf16* b1 = bch + (size_t)(n0 + 16 + r) * TDM_KC + ks + half * 8;
      v16bf bb1 = pack16(*(const v8bf*)b1, *(const v8bf*)(b1 + 16));
      acc1 = __builtin_amdgcn_wmma_f32_16x16x32_bf16(
          false, a, false, bb1, (short)0, acc1, false, false);
    }
    __syncthreads();                                   // buffer reuse fence
    buf ^= 1;
  }

#pragma unroll
  for (int v = 0; v < 8; ++v) {
    int row = m0 + v + 8 * half;
    float v0 = acc0[v], v1 = acc1[v];
    v0 = v0 > 0.f ? v0 : (__expf(v0) - 1.f);           // ELU
    v1 = v1 > 0.f ? v1 : (__expf(v1) - 1.f);
    Cf[(size_t)row * HDIM + n0 + r]      = v0;
    Cf[(size_t)row * HDIM + n0 + 16 + r] = v1;
  }
}

// xf[i] = leaky( dot(T3[i,0:128], w[0:128]) )
__global__ __launch_bounds__(256) void matvec_leaky(const float* __restrict__ T3,
                                                    const float* __restrict__ w,
                                                    float* __restrict__ xf, int M) {
  int i = blockIdx.x * blockDim.x + threadIdx.x;
  if (i >= M) return;
  const float* row = T3 + (size_t)i * 128;
  float s = 0.f;
#pragma unroll 4
  for (int k = 0; k < 128; ++k) s += row[k] * w[k];
  xf[i] = s > 0.f ? s : 0.01f * s;
}

// exact GELU + cubic B-spline bases on grid linspace(-2.2, 2.2, 12)
__global__ __launch_bounds__(256) void spline_kernel(const float* __restrict__ xf,
                                                     float* __restrict__ base,
                                                     float* __restrict__ bases, int n) {
  int i = blockIdx.x * blockDim.x + threadIdx.x;
  if (i >= n) return;
  float x = xf[i];
  base[i] = 0.5f * x * (1.f + erff(x * 0.7071067811865476f));
  float g[12];
#pragma unroll
  for (int t = 0; t < 12; ++t) g[t] = -2.2f + 0.4f * (float)t;
  float b[11];
#pragma unroll
  for (int t = 0; t < 11; ++t) b[t] = (x >= g[t] && x < g[t + 1]) ? 1.f : 0.f;
#pragma unroll
  for (int k = 1; k <= 3; ++k) {
    for (int t = 0; t < 11 - k; ++t) {   // in-place forward: reads old b[t], b[t+1]
      float left  = (x - g[t]) / (g[t + k] - g[t]) * b[t];
      float right = (g[t + k + 1] - x) / (g[t + k + 1] - g[t + 1]) * b[t + 1];
      b[t] = left + right;
    }
  }
#pragma unroll
  for (int j = 0; j < 8; ++j) bases[(size_t)i * 8 + j] = b[j];
}

// One block per channel c: conv_cn[c,s] = base_w[c]*base[s] + spline_w[c,:]·bases[s,:]
// then InstanceNorm (biased var, eps 1e-5) + PReLU. torch .view is a flat
// reinterpretation, so hc (row-major [4096,64]) == flat [64*4096] buffer.
// Also emits bf16 transpose hcT[64,4096] for the attention WMMA GEMM.
__global__ __launch_bounds__(256) void conv_norm_kernel(
    const float* __restrict__ base, const float* __restrict__ bases,
    const float* __restrict__ base_w, const float* __restrict__ spline_w,
    const float* __restrict__ prelu_p, float* __restrict__ hc,
    __bf16* __restrict__ hcT, int Ns) {
  __shared__ float vals[NODES];
  __shared__ float red[16];
  int c = blockIdx.x;
  float bw = base_w[c];
  float sw[8];
#pragma unroll
  for (int j = 0; j < 8; ++j) sw[j] = spline_w[c * 8 + j];
  float sum = 0.f, sumsq = 0.f;
  for (int s = threadIdx.x; s < Ns; s += blockDim.x) {
    const float* bp = bases + (size_t)s * 8;
    float v = bw * base[s];
#pragma unroll
    for (int j = 0; j < 8; ++j) v += sw[j] * bp[j];
    vals[s] = v; sum += v; sumsq += v * v;
  }
  for (int off = 16; off; off >>= 1) {
    sum   += __shfl_xor(sum, off, 32);
    sumsq += __shfl_xor(sumsq, off, 32);
  }
  int wv = threadIdx.x >> 5;
  if ((threadIdx.x & 31) == 0) { red[wv] = sum; red[8 + wv] = sumsq; }
  __syncthreads();
  if (threadIdx.x == 0) {
    float s0 = 0.f, s1 = 0.f;
    for (int q = 0; q < 8; ++q) { s0 += red[q]; s1 += red[8 + q]; }
    red[0] = s0; red[8] = s1;
  }
  __syncthreads();
  float mu   = red[0] / (float)Ns;
  float var  = red[8] / (float)Ns - mu * mu;
  float rstd = rsqrtf(var + 1e-5f);
  float p = prelu_p[0];
  for (int s = threadIdx.x; s < Ns; s += blockDim.x) {
    float v = (vals[s] - mu) * rstd;
    v = v > 0.f ? v : p * v;
    size_t f = (size_t)c * Ns + s;
    hc[f] = v;
    hcT[(size_t)(f & 63) * Ns + (f >> 6)] = f2bf(v);
  }
}

__global__ __launch_bounds__(256) void wh_kernel(const float* __restrict__ hc,
                                                 const float* __restrict__ a,
                                                 float* __restrict__ wh1,
                                                 float* __restrict__ wh2, int Nn) {
  int n = blockIdx.x * blockDim.x + threadIdx.x;
  if (n >= Nn) return;
  const float* row = hc + (size_t)n * HDIM;
  float s1 = 0.f, s2 = 0.f;
#pragma unroll 4
  for (int d = 0; d < HDIM; ++d) { s1 += row[d] * a[d]; s2 += row[d] * a[HDIM + d]; }
  wh1[n] = s1; wh2[n] = s2;
}

// One block per attention row: masked ELU -> softmax -> bf16 row.
// adj row streamed from HBM exactly once (dominant traffic, 64MB total).
__global__ __launch_bounds__(256) void att_row_kernel(
    const float* __restrict__ adj, const float* __restrict__ wh1,
    const float* __restrict__ wh2, __bf16* __restrict__ attc, int row0, int Nn) {
  __shared__ float vals[NODES];
  __shared__ float red[8];
  int i = row0 + blockIdx.x;
  float w1 = wh1[i];
  const float* arow = adj + (size_t)i * Nn;
  float mx = -3.4e38f;
  for (int j = threadIdx.x; j < Nn; j += blockDim.x) {
    float v = w1 + wh2[j];
    v = v > 0.f ? v : (__expf(v) - 1.f);          // ELU
    v = arow[j] > 0.f ? v : -9.0e15f;             // mask
    vals[j] = v; mx = fmaxf(mx, v);
  }
  for (int off = 16; off; off >>= 1) mx = fmaxf(mx, __shfl_xor(mx, off, 32));
  if ((threadIdx.x & 31) == 0) red[threadIdx.x >> 5] = mx;
  __syncthreads();
  if (threadIdx.x == 0) {
    float t = red[0];
    for (int q = 1; q < 8; ++q) t = fmaxf(t, red[q]);
    red[0] = t;
  }
  __syncthreads();
  float m = red[0];
  __syncthreads();
  float sum = 0.f;
  for (int j = threadIdx.x; j < Nn; j += blockDim.x) {
    float pv = __expf(vals[j] - m);
    vals[j] = pv; sum += pv;
  }
  for (int off = 16; off; off >>= 1) sum += __shfl_xor(sum, off, 32);
  if ((threadIdx.x & 31) == 0) red[threadIdx.x >> 5] = sum;
  __syncthreads();
  if (threadIdx.x == 0) {
    float t = 0.f;
    for (int q = 0; q < 8; ++q) t += red[q];
    red[0] = t;
  }
  __syncthreads();
  float inv = 1.f / red[0];
  __bf16* orow = attc + (size_t)blockIdx.x * Nn;
  for (int j = threadIdx.x; j < Nn; j += blockDim.x)
    orow[j] = f2bf(vals[j] * inv);
}

// embed = F.normalize(hp) + conv_bias  (hp already has ELU from gemm epilogue)
__global__ __launch_bounds__(256) void embed_kernel(const float* __restrict__ hp,
                                                    const float* __restrict__ cbias,
                                                    __bf16* __restrict__ embed_bf, int Nn) {
  int n = blockIdx.x * blockDim.x + threadIdx.x;
  if (n >= Nn) return;
  const float* row = hp + (size_t)n * HDIM;
  float ss = 0.f;
#pragma unroll 4
  for (int d = 0; d < HDIM; ++d) ss += row[d] * row[d];
  float inv = 1.f / fmaxf(sqrtf(ss), 1e-12f);
#pragma unroll 4
  for (int d = 0; d < HDIM; ++d)
    embed_bf[(size_t)n * HDIM + d] = f2bf(row[d] * inv + cbias[d]);
}

__global__ __launch_bounds__(256) void decode_kernel(const int* __restrict__ ts,
                                                     const float* __restrict__ tf,
                                                     const float* __restrict__ tg,
                                                     float* __restrict__ out, int E) {
  int e = blockIdx.x * blockDim.x + threadIdx.x;
  if (e >= E) return;
  int i = ts[2 * e], j = ts[2 * e + 1];
  const float4* a = (const float4*)(tf + (size_t)i * 16);
  const float4* b = (const float4*)(tg + (size_t)j * 16);
  float s = 0.f;
#pragma unroll
  for (int q = 0; q < 4; ++q) {
    float4 x = a[q], y = b[q];
    s += x.x * y.x + x.y * y.y + x.z * y.z + x.w * y.w;
  }
  out[e] = s;
}

// ---------------------------------------------------------------------------
extern "C" void kernel_launch(void* const* d_in, const int* in_sizes, int n_in,
                              void* d_out, int out_size, void* d_ws, size_t ws_size,
                              hipStream_t stream) {
  const float* x        = (const float*)d_in[0];
  const float* adj      = (const float*)d_in[1];
  const int*   ts       = (const int*)  d_in[2];
  const float* g1_w_in  = (const float*)d_in[3];
  const float* g1_w_out = (const float*)d_in[4];
  const float* g2_w_in  = (const float*)d_in[5];
  const float* g2_w_out = (const float*)d_in[6];
  const float* g3_w_in  = (const float*)d_in[7];
  const float* g3_w_out = (const float*)d_in[8];
  const float* base_w   = (const float*)d_in[9];
  const float* spline_w = (const float*)d_in[10];
  const float* prelu_p  = (const float*)d_in[11];
  const float* attn_a   = (const float*)d_in[12];
  const float* conv_b   = (const float*)d_in[13];
  const float* tf_w1 = (const float*)d_in[14]; const float* tf_b1 = (const float*)d_in[15];
  const float* tf_w2 = (const float*)d_in[16]; const float* tf_b2 = (const float*)d_in[17];
  const float* tg_w1 = (const float*)d_in[18]; const float* tg_b1 = (const float*)d_in[19];
  const float* tg_w2 = (const float*)d_in[20]; const float* tg_b2 = (const float*)d_in[21];
  float* out = (float*)d_out;
  const int E = out_size;        // 131072 edges
  const int Nn = NODES;

  // ---- workspace carving (~18 MB, 256B aligned slabs) ----
  char* w = (char*)d_ws;
  size_t off = 0;
  auto alloc = [&](size_t bytes) -> void* {
    void* p = w + off;
    off = (off + bytes + 255) & ~(size_t)255;
    return p;
  };
  __bf16* x_bf   = (__bf16*)alloc((size_t)Nn * INDIM * 2);
  __bf16* w1i_bf = (__bf16*)alloc(128 * INDIM * 2);
  __bf16* w1o_bf = (__bf16*)alloc(64 * 128 * 2);
  __bf16* w2i_bf = (__bf16*)alloc(128 * 64 * 2);
  __bf16* w2o_bf = (__bf16*)alloc(64 * 128 * 2);
  __bf16* w3i_bf = (__bf16*)alloc(128 * 64 * 2);
  __bf16* tfw1_bf = (__bf16*)alloc(64 * 64 * 2);
  __bf16* tfw2_bf = (__bf16*)alloc(16 * 64 * 2);
  __bf16* tgw1_bf = (__bf16*)alloc(64 * 64 * 2);
  __bf16* tgw2_bf = (__bf16*)alloc(16 * 64 * 2);
  __bf16* T1_bf  = (__bf16*)alloc((size_t)Nn * 128 * 2);
  __bf16* H1_bf  = (__bf16*)alloc((size_t)Nn * 64 * 2);
  __bf16* T2_bf  = (__bf16*)alloc((size_t)Nn * 128 * 2);
  __bf16* H2_bf  = (__bf16*)alloc((size_t)Nn * 64 * 2);
  float*  T3_f   = (float*) alloc((size_t)Nn * 128 * 4);
  float*  xf     = (float*) alloc((size_t)Nn * 4);
  float*  base   = (float*) alloc((size_t)Nn * 4);
  float*  bases  = (float*) alloc((size_t)Nn * 8 * 4);
  float*  hc_f   = (float*) alloc((size_t)64 * Nn * 4);
  __bf16* hcT_bf = (__bf16*)alloc((size_t)64 * Nn * 2);
  float*  wh1    = (float*) alloc((size_t)Nn * 4);
  float*  wh2    = (float*) alloc((size_t)Nn * 4);
  const int CH = 512;  // attention row chunk
  __bf16* att_bf = (__bf16*)alloc((size_t)CH * Nn * 2);
  float*  hp_f   = (float*) alloc((size_t)Nn * 64 * 4);
  __bf16* emb_bf = (__bf16*)alloc((size_t)Nn * 64 * 2);
  __bf16* tf1_bf = (__bf16*)alloc((size_t)Nn * 64 * 2);
  __bf16* tg1_bf = (__bf16*)alloc((size_t)Nn * 64 * 2);
  float*  tf_f   = (float*) alloc((size_t)Nn * 16 * 4);
  float*  tg_f   = (float*) alloc((size_t)Nn * 16 * 4);
  (void)ws_size; (void)n_in; (void)in_sizes;

  auto cdiv = [](int a, int b) { return (a + b - 1) / b; };
  auto cvt = [&](const float* s, __bf16* d, int n) {
    cvt_f32_bf16<<<cdiv(n, 256), 256, 0, stream>>>(s, d, n);
  };
  // 2-wide N-blocked gemm for all N%32==0 cases
  auto gemm2 = [&](const __bf16* A, const __bf16* B, float* Cf, __bf16* Ch,
                   const float* bias, int M, int N, int K, int act) {
    int tiles = (M >> 4) * (N >> 5);
    wmma_gemm_nt<2><<<cdiv(tiles, 8), 256, 0, stream>>>(A, B, Cf, Ch, bias, M, N, K, act);
  };
  auto gemm1 = [&](const __bf16* A, const __bf16* B, float* Cf, __bf16* Ch,
                   const float* bias, int M, int N, int K, int act) {
    int tiles = (M >> 4) * (N >> 4);
    wmma_gemm_nt<1><<<cdiv(tiles, 8), 256, 0, stream>>>(A, B, Cf, Ch, bias, M, N, K, act);
  };

  // ---- bf16 operand prep ----
  cvt(x, x_bf, Nn * INDIM);
  cvt(g1_w_in, w1i_bf, 128 * INDIM);
  cvt(g1_w_out, w1o_bf, 64 * 128);
  cvt(g2_w_in, w2i_bf, 128 * 64);
  cvt(g2_w_out, w2o_bf, 64 * 128);
  cvt(g3_w_in, w3i_bf, 128 * 64);
  cvt(tf_w1, tfw1_bf, 64 * 64);
  cvt(tf_w2, tfw2_bf, 16 * 64);
  cvt(tg_w1, tgw1_bf, 64 * 64);
  cvt(tg_w2, tgw2_bf, 16 * 64);

  // ---- three KanGNN blocks ----
  gemm2(x_bf,  w1i_bf, nullptr, T1_bf, nullptr, Nn, 128, INDIM, 0);
  gemm2(T1_bf, w1o_bf, nullptr, H1_bf, nullptr, Nn, 64, 128, 1);   // leaky
  gemm2(H1_bf, w2i_bf, nullptr, T2_bf, nullptr, Nn, 128, 64, 0);
  gemm2(T2_bf, w2o_bf, nullptr, H2_bf, nullptr, Nn, 64, 128, 1);   // leaky
  gemm2(H2_bf, w3i_bf, T3_f, nullptr, nullptr, Nn, 128, 64, 0);
  matvec_leaky<<<cdiv(Nn, 256), 256, 0, stream>>>(T3_f, g3_w_out, xf, Nn);

  // ---- KANConv2D + InstanceNorm + PReLU ----
  spline_kernel<<<cdiv(Nn, 256), 256, 0, stream>>>(xf, base, bases, Nn);
  conv_norm_kernel<<<64, 256, 0, stream>>>(base, bases, base_w, spline_w,
                                           prelu_p, hc_f, hcT_bf, Nn);

  // ---- GAT attention ----
  wh_kernel<<<cdiv(Nn, 256), 256, 0, stream>>>(hc_f, attn_a, wh1, wh2, Nn);
  for (int row0 = 0; row0 < Nn; row0 += CH) {
    att_row_kernel<<<CH, 256, 0, stream>>>(adj, wh1, wh2, att_bf, row0, Nn);
    // hp[row0:row0+CH,:] = elu(att_chunk @ hcT^T), B staged via TDM
    wmma_gemm_att_tdm<<<CH / 64, 256, 0, stream>>>(
        att_bf, hcT_bf, hp_f + (size_t)row0 * HDIM, Nn);
  }
  embed_kernel<<<cdiv(Nn, 256), 256, 0, stream>>>(hp_f, conv_b, emb_bf, Nn);

  // ---- tf / tg heads ----
  gemm2(emb_bf, tfw1_bf, nullptr, tf1_bf, tf_b1, Nn, 64, 64, 0);
  gemm1(tf1_bf, tfw2_bf, tf_f, nullptr, tf_b2, Nn, 16, 64, 0);
  gemm2(emb_bf, tgw1_bf, nullptr, tg1_bf, tg_b1, Nn, 64, 64, 0);
  gemm1(tg1_bf, tgw2_bf, tg_f, nullptr, tg_b2, Nn, 16, 64, 0);

  // ---- decode ----
  decode_kernel<<<cdiv(E, 256), 256, 0, stream>>>(ts, tf_f, tg_f, out, E);
}
